// DynamicConv2D_88519275970729
// MI455X (gfx1250) — matrix operand order repaired and checked by reading, not compiled, not run on hardware
//
#include <hip/hip_runtime.h>

typedef __attribute__((ext_vector_type(16))) _Float16 v16h;
typedef __attribute__((ext_vector_type(8)))  float    v8f;

#define BATCH 32
#define CIN   16
#define COUT  16
#define HH    256
#define WW    256
#define KTOT  144          // Cin*3*3
#define KCH   5            // 144 padded to 160 = 5 chunks of K=32
#define TILE_W 16
#define TILE_H 32          // 4 rows per wave x 8 waves
#define RPW    4           // rows (accumulators) per wave
#define TROWS  35          // TILE_H + 2 halo + 1 row reachable by K-padding tap
#define TCOLS  18          // TILE_W + 2 halo
#define ROWSTRIDE (TCOLS * CIN)        // halfs per tile row
#define WAVESTEP  (8 * ROWSTRIDE)      // 8 rows between a wave's accumulators

// LDS half-offset of tap group g (g = ky*3+kx, 0..9): (ky*TCOLS + kx) * CIN
#define GOFF(g) ((((g) / 3) * TCOLS + ((g) % 3)) * CIN)

__global__ __launch_bounds__(256)
void dynconv2d_wmma_f16(const float* __restrict__ x,
                        const float* __restrict__ w,
                        float* __restrict__ out)
{
    // Channel-LAST input tile: 16 channels of one (row,col) point are 32
    // contiguous bytes == exactly one v16h WMMA-B half-wave fragment.
    __shared__ __align__(32) _Float16 Xlds[TROWS][TCOLS][CIN];
    // Weights pre-swizzled into the WMMA 16-bit A per-lane layout, K-permuted
    // to k' = (ky*3+kx)*16 + ci, zero-padded 144 -> 160.
    __shared__ __align__(32) _Float16 Alds[KCH][32][16];

    const int tid  = threadIdx.x;
    const int lane = tid & 31;
    const int wid  = tid >> 5;           // wave id; wave rows = wid + 8*{0..3}
    const int g    = blockIdx.z;         // group
    const int x0   = blockIdx.x * TILE_W;
    const int y0   = blockIdx.y * TILE_H;

    // ---- stage input tile (f32 -> f16), rows y0-1 .. y0+33, cols x0-1 .. x0+16
    // Global reads col-fastest (coalesced); LDS writes channel-last scatter.
    const float* xg = x + (size_t)g * CIN * HH * WW;
    for (int i = tid; i < CIN * TROWS * TCOLS; i += 256) {
        int ci = i / (TROWS * TCOLS);
        int r  = (i / TCOLS) % TROWS;
        int cc = i % TCOLS;
        int gy = y0 - 1 + r;
        int gx = x0 - 1 + cc;
        float v = 0.f;
        if ((unsigned)gy < (unsigned)HH && (unsigned)gx < (unsigned)WW)
            v = xg[(size_t)ci * HH * WW + gy * WW + gx];
        Xlds[r][cc][ci] = (_Float16)v;
    }

    // ---- stage weights into A-fragment layout (permuted K) ----
    // lane<16: M=lane,    halves h hold k'_local = {0..15} (+32c)
    // lane>=16: M=lane-16, same +16.  Original k = ci*9 + gidx.
    const float* wg = w + (size_t)g * COUT * KTOT;
    if (tid < KCH * 32) {
        int c  = tid >> 5;
        int l  = tid & 31;
        int m  = l & 15;
        int kb = (l < 16) ? 0 : 16;
        v16h av;
        #pragma unroll
        for (int h = 0; h < 16; ++h) {
            int klocal = h + kb;
            int kp   = 32 * c + klocal;      // permuted K index
            int gidx = kp >> 4;              // tap group 0..9
            int ci   = kp & 15;              // channel
            float v = (gidx < 9) ? wg[m * KTOT + ci * 9 + gidx] : 0.f;
            av[h] = (_Float16)v;
        }
        *(v16h*)&Alds[c][l][0] = av;         // 2x ds_store_b128
    }

    __syncthreads();

    // ---- GEMM: 4 accumulators per wave share each A-fragment ----
    const int N  = lane & 15;                // pixel column within tile
    const int hs = lane >> 4;                // half-wave select (B K-base 0/16)

    // Per-lane fragment base at (row=wid, col=N, ch=0); the wave's other rows
    // are +8/+16/+24 rows away == compile-time ds immediate offsets.
    const _Float16* bbase = &Xlds[0][0][0] + (wid * TCOLS + N) * CIN;

    v8f acc0 = {}, acc1 = {}, acc2 = {}, acc3 = {};
    #pragma unroll
    for (int c = 0; c < KCH; ++c) {
        v16h a = *(const v16h*)&Alds[c][lane][0];          // 2x ds_load_b128
        // tap group: 2c for lanes 0-15, 2c+1 for lanes 16-31 (cndmask of imms)
        int goff = hs ? GOFF(2 * c + 1) : GOFF(2 * c);
        const _Float16* bp = bbase + goff;
        v16h b0 = *(const v16h*)(bp);                      // rows wid + 8k
        v16h b1 = *(const v16h*)(bp + 1 * WAVESTEP);
        v16h b2 = *(const v16h*)(bp + 2 * WAVESTEP);
        v16h b3 = *(const v16h*)(bp + 3 * WAVESTEP);
        acc0 = __builtin_amdgcn_wmma_f32_16x16x32_f16(false, a, false, b0,
                                                      (short)0, acc0, false, false);
        acc1 = __builtin_amdgcn_wmma_f32_16x16x32_f16(false, a, false, b1,
                                                      (short)0, acc1, false, false);
        acc2 = __builtin_amdgcn_wmma_f32_16x16x32_f16(false, a, false, b2,
                                                      (short)0, acc2, false, false);
        acc3 = __builtin_amdgcn_wmma_f32_16x16x32_f16(false, a, false, b3,
                                                      (short)0, acc3, false, false);
    }

    // ---- store: VGPR v -> M = v + (lane<16 ? 0 : 8), N = lane%16 ----
    const int mbase = hs ? 8 : 0;
    float* og = out + (size_t)g * COUT * HH * WW
                    + (size_t)(y0 + wid) * WW + x0 + N;
    #pragma unroll
    for (int v = 0; v < 8; ++v) {
        size_t mo = (size_t)(mbase + v) * HH * WW;
        og[mo            ] = acc0[v];
        og[mo +  8 * WW  ] = acc1[v];
        og[mo + 16 * WW  ] = acc2[v];
        og[mo + 24 * WW  ] = acc3[v];
    }
}

extern "C" void kernel_launch(void* const* d_in, const int* in_sizes, int n_in,
                              void* d_out, int out_size, void* d_ws, size_t ws_size,
                              hipStream_t stream) {
    const float* x = (const float*)d_in[0];      // [1, 32*16, 256, 256] f32
    const float* w = (const float*)d_in[1];      // [32*16, 16, 3, 3]    f32
    float* out = (float*)d_out;                  // [1, 32*16, 256, 256] f32
    (void)in_sizes; (void)n_in; (void)out_size; (void)d_ws; (void)ws_size;

    dim3 grid(WW / TILE_W, HH / TILE_H, BATCH);  // 16 x 8 x 32 blocks
    dynconv2d_wmma_f16<<<grid, 256, 0, stream>>>(x, w, out);
}
